// DeepConvLSTM_16071767622010
// MI455X (gfx1250) — compile-verified, matrix operand order checked by reading, not run on hardware
//
#include <hip/hip_runtime.h>
#include <hip/hip_bf16.h>
#include <stdint.h>

typedef __attribute__((ext_vector_type(16))) _Float16 v16h;
typedef __attribute__((ext_vector_type(8)))  _Float16 v8h;
typedef __attribute__((ext_vector_type(8)))  float    v8f;

#define B_     8
#define T_     20
#define CIN_   2
#define HID_   128
#define HW_    42
#define P_     (HW_*HW_)      // 1764 pixels
#define NGATE_ (4*HID_)       // 512
#define EPS_   1e-5f

// ---------------------------------------------------------------------------
// 1) Repack conv weights OIHW fp32 -> tap-major, K-contiguous f16
//    [25][512][Cpad].  Cpad = roundup32(Ctot), zero-filled pad channels.
// ---------------------------------------------------------------------------
__global__ void wconv_kernel(const float* __restrict__ w, _Float16* __restrict__ wt,
                             int ctot, int cpad) {
    int idx = blockIdx.x * 256 + threadIdx.x;       // over 25*512*cpad
    if (idx >= 25 * NGATE_ * cpad) return;
    int ci = idx % cpad;
    int n  = (idx / cpad) % NGATE_;
    int r  = idx / (cpad * NGATE_);                 // tap 0..24
    float v = 0.f;
    if (ci < ctot) v = w[((n * ctot) + ci) * 25 + r];
    wt[idx] = (_Float16)v;
}

// ---------------------------------------------------------------------------
// 2) Build combined f16 input, channel-innermost: [b][P][Cpad].
//    ci < cin_l : x (layer 0) or previous layer's h_t; then HID channels of
//    own h_{t-1}; then zero pad to Cpad (removes all bounds logic in conv).
// ---------------------------------------------------------------------------
__global__ void pack_comb_kernel(const float* __restrict__ x,
                                 const _Float16* __restrict__ seq_prev,
                                 const _Float16* __restrict__ seq_cur,
                                 _Float16* __restrict__ comb,
                                 int cin_l, int cpad, int t) {
    int idx = blockIdx.x * 256 + threadIdx.x;
    int tot = B_ * P_ * cpad;
    if (idx >= tot) return;
    int ci = idx % cpad;
    int p  = (idx / cpad) % P_;
    int b  = idx / (cpad * P_);
    _Float16 v = (_Float16)0.f;
    if (ci < cin_l) {
        if (x) v = (_Float16)x[(((size_t)(b * T_ + t)) * cin_l + ci) * P_ + p];
        else   v = seq_prev[(((size_t)(b * T_ + t)) * HID_ + ci) * P_ + p];
    } else if (ci < cin_l + HID_) {
        int hc = ci - cin_l;
        if (t > 0) v = seq_cur[(((size_t)(b * T_ + (t - 1))) * HID_ + hc) * P_ + p];
    }
    comb[idx] = v;
}

// ---------------------------------------------------------------------------
// 3) Tap-major implicit-GEMM 5x5 SAME conv via WMMA f16->f32 with
//    ASYNC global->LDS staging (gfx1250 GLOBAL_LOAD_ASYNC_TO_LDS_B128,
//    tracked by ASYNCcnt; closed with s_wait_asynccnt before the barrier).
//    Block = 128 threads (4 waves), 64(M pixels) x 128(N channels) tile;
//    each wave owns a 32x64 quadrant = 2x4 accumulators = 8 WMMA per K-step.
//    gates layout: [b][P][512]  (coalesced WMMA stores).
// ---------------------------------------------------------------------------
__global__ __launch_bounds__(128)
void conv_wmma_kernel(const _Float16* __restrict__ comb,
                      const _Float16* __restrict__ wt,
                      float* __restrict__ gates,
                      int cpad) {
    // Row stride 40 halves = 20 dwords; gcd(20,64)=4 -> 16 lanes at stride 20
    // cover 16 distinct banks -> conflict-free ds_load_b128 fragment reads.
    // Rows are 16B-aligned (80B stride), as async b128 and v8h stores need.
    __shared__ __align__(16) _Float16 At[64][40];
    __shared__ __align__(16) _Float16 Bt[128][40];

    const int tid  = threadIdx.x;
    const int wave = tid >> 5, lane = tid & 31;
    const int hb   = lane >> 4, lr = lane & 15;     // K-half select / row-in-16
    const int mh   = wave & 1,  nh = wave >> 1;     // wave quadrant (32x64)
    const int mBase = blockIdx.x * 64;
    const int nBase = blockIdx.y * 128;
    const int b     = blockIdx.z;
    const _Float16* combB = comb + (size_t)b * P_ * cpad;

    // A-fill geometry: fixed for entire kernel
    const int  arow    = tid & 63;                  // mi (row in A tile)
    const int  akh     = (tid >> 6) * 16;           // ki base: 0 or 16
    const int  apix    = mBase + arow;
    const int  aoh     = apix / HW_, aow = apix % HW_;   // once per thread
    const bool apix_ok = apix < P_;

    const unsigned adst = (unsigned)(size_t)&At[arow][akh];  // LDS byte offset
    const unsigned bdst = (unsigned)(size_t)&Bt[tid][0];

    v8f acc[2][4];
#pragma unroll
    for (int i = 0; i < 2; ++i)
#pragma unroll
        for (int j = 0; j < 4; ++j) acc[i][j] = (v8f){0,0,0,0,0,0,0,0};

    for (int r = 0; r < 25; ++r) {
        const int dh = r / 5 - 2, dw = r % 5 - 2;
        const int ih = aoh + dh, iw = aow + dw;
        const bool avalid = apix_ok && (unsigned)ih < HW_ && (unsigned)iw < HW_;
        // shifted-pixel row of the K-contiguous activation matrix
        const _Float16* arowp = combB + (size_t)(ih * HW_ + iw) * cpad + akh;
        // this thread's n-row of the K-contiguous weight matrix
        const _Float16* browp = wt + ((size_t)r * NGATE_ + nBase + tid) * cpad;

        for (int kc = 0; kc < cpad; kc += 32) {
            // ---- async stage A: 32B of one shifted pixel row (or zeros) ----
            if (avalid) {
                const _Float16* as = arowp + kc;
                asm volatile("global_load_async_to_lds_b128 %0, %1, off"
                             :: "v"(adst), "v"(as) : "memory");
                asm volatile("global_load_async_to_lds_b128 %0, %1, off"
                             :: "v"(adst + 16), "v"(as + 8) : "memory");
            } else {
                v8h z = (v8h){0,0,0,0,0,0,0,0};
                *(v8h*)&At[arow][akh]     = z;
                *(v8h*)&At[arow][akh + 8] = z;
            }
            // ---- async stage B: 64B = one weight n-row of this K-block ----
            {
                const _Float16* bs = browp + kc;
#pragma unroll
                for (int q = 0; q < 4; ++q)
                    asm volatile("global_load_async_to_lds_b128 %0, %1, off"
                                 :: "v"(bdst + 16 * q), "v"(bs + 8 * q) : "memory");
            }
            asm volatile("s_wait_asynccnt 0x0" ::: "memory");
            __syncthreads();

            // ---- fragments: ISA 16-bit A/B layout.
            // lane half hb: element e holds K = (e<8 ? e : 8+e) + 8*hb
            // -> two contiguous 16B chunks per row: [8hb..+7], [16+8hb..+7]
            v16h afr[2], bfr[4];
#pragma unroll
            for (int s = 0; s < 2; ++s) {
                int m = mh * 32 + s * 16 + lr;
                v8h lo = *(const v8h*)&At[m][8 * hb];
                v8h hi = *(const v8h*)&At[m][16 + 8 * hb];
#pragma unroll
                for (int e = 0; e < 8; ++e) { afr[s][e] = lo[e]; afr[s][8 + e] = hi[e]; }
            }
#pragma unroll
            for (int s = 0; s < 4; ++s) {
                int n = nh * 64 + s * 16 + lr;
                v8h lo = *(const v8h*)&Bt[n][8 * hb];
                v8h hi = *(const v8h*)&Bt[n][16 + 8 * hb];
#pragma unroll
                for (int e = 0; e < 8; ++e) { bfr[s][e] = lo[e]; bfr[s][8 + e] = hi[e]; }
            }
#pragma unroll
            for (int i = 0; i < 2; ++i)
#pragma unroll
                for (int j = 0; j < 4; ++j)
                    acc[i][j] = __builtin_amdgcn_wmma_f32_16x16x32_f16(
                        false, afr[i], false, bfr[j], (short)0, acc[i][j], false, false);
            __syncthreads();
        }
    }

    // ---- store: C/D layout lane%16 -> N, VGPR v -> M = v + 8*(lane/16) ----
    float* gb = gates + (size_t)b * P_ * NGATE_;
#pragma unroll
    for (int i = 0; i < 2; ++i) {
#pragma unroll
        for (int j = 0; j < 4; ++j) {
            int n = nBase + nh * 64 + j * 16 + lr;
#pragma unroll
            for (int v = 0; v < 8; ++v) {
                int m = mBase + mh * 32 + i * 16 + v + 8 * hb;
                if (m < P_) gb[(size_t)m * NGATE_ + n] = acc[i][j][v];
            }
        }
    }
}

// ---------------------------------------------------------------------------
// 4) LSTM pointwise: gates -> c update, pre-LayerNorm h.  c/pre: [b][hc][P]
// ---------------------------------------------------------------------------
__global__ void lstm_pointwise_kernel(const float* __restrict__ gates,
                                      const float* __restrict__ cb,
                                      const float* __restrict__ wci,
                                      const float* __restrict__ wcf,
                                      const float* __restrict__ wco,
                                      float* __restrict__ cbuf,
                                      float* __restrict__ pre) {
    int idx = blockIdx.x * 256 + threadIdx.x;
    int tot = B_ * HID_ * P_;
    if (idx >= tot) return;
    int p  = idx % P_;
    int hc = (idx / P_) % HID_;
    int b  = idx / (P_ * HID_);
    const float* g = gates + ((size_t)b * P_ + p) * NGATE_;
    float gi = g[hc]            + cb[hc];
    float gf = g[HID_ + hc]     + cb[HID_ + hc];
    float gg = g[2 * HID_ + hc] + cb[2 * HID_ + hc];
    float go = g[3 * HID_ + hc] + cb[3 * HID_ + hc];
    float c  = cbuf[idx];
    float i_ = 1.f / (1.f + __expf(-(gi + wci[hc] * c)));
    float f_ = 1.f / (1.f + __expf(-(gf + wcf[hc] * c)));
    float g_ = tanhf(gg);
    float cn = f_ * c + i_ * g_;
    float o_ = 1.f / (1.f + __expf(-(go + wco[hc] * cn)));
    cbuf[idx] = cn;
    pre[idx]  = o_ * tanhf(cn);
}

// ---------------------------------------------------------------------------
// 5) Per-batch-sample mean / rstd over (C,H,W)
// ---------------------------------------------------------------------------
__global__ void stats_kernel(const float* __restrict__ pre, float* __restrict__ stats) {
    __shared__ float ssum[256], ssq[256];
    int b = blockIdx.x;
    const float* pb = pre + (size_t)b * HID_ * P_;
    float s = 0.f, q = 0.f;
    for (int i = threadIdx.x; i < HID_ * P_; i += 256) {
        float v = pb[i]; s += v; q += v * v;
    }
    ssum[threadIdx.x] = s; ssq[threadIdx.x] = q;
    __syncthreads();
    for (int st = 128; st > 0; st >>= 1) {
        if (threadIdx.x < st) {
            ssum[threadIdx.x] += ssum[threadIdx.x + st];
            ssq[threadIdx.x]  += ssq[threadIdx.x + st];
        }
        __syncthreads();
    }
    if (threadIdx.x == 0) {
        float n  = (float)(HID_ * P_);
        float mu = ssum[0] / n;
        float var = ssq[0] / n - mu * mu;
        stats[b * 2]     = mu;
        stats[b * 2 + 1] = rsqrtf(var + EPS_);
    }
}

// ---------------------------------------------------------------------------
// 6) LayerNorm -> f16 sequence buffer (+ fp32 d_out for last layer)
// ---------------------------------------------------------------------------
__global__ void norm_kernel(const float* __restrict__ pre,
                            const float* __restrict__ stats,
                            const float* __restrict__ lnw,
                            const float* __restrict__ lnb,
                            _Float16* __restrict__ seq_cur,
                            float* __restrict__ dout, int t) {
    int idx = blockIdx.x * 256 + threadIdx.x;
    int tot = B_ * HID_ * P_;
    if (idx >= tot) return;
    int p  = idx % P_;
    int hc = (idx / P_) % HID_;
    int b  = idx / (P_ * HID_);
    float mu = stats[b * 2], rs = stats[b * 2 + 1];
    float h  = (pre[idx] - mu) * rs * lnw[hc * P_ + p] + lnb[hc * P_ + p];
    size_t o = (((size_t)(b * T_ + t)) * HID_ + hc) * P_ + p;
    seq_cur[o] = (_Float16)h;
    if (dout) dout[o] = h;
}

__global__ void zero_kernel(float* __restrict__ buf, int n) {
    int idx = blockIdx.x * 256 + threadIdx.x;
    if (idx < n) buf[idx] = 0.f;
}

// ---------------------------------------------------------------------------
// 7) Final sigmoid attention on t = T-1 frame, in-place in d_out
// ---------------------------------------------------------------------------
__global__ void attn_kernel(float* __restrict__ dout,
                            const float* __restrict__ attn_w,
                            const float* __restrict__ attn_b) {
    __shared__ float red[128];
    int bp = blockIdx.x;            // over B_*P_
    int b = bp / P_, p = bp % P_;
    int c = threadIdx.x;            // 128 channels
    float* base = dout + (((size_t)(b * T_ + (T_ - 1))) * HID_) * P_;
    float v = base[(size_t)c * P_ + p];
    red[c] = v * attn_w[c];
    __syncthreads();
    for (int st = 64; st > 0; st >>= 1) {
        if (c < st) red[c] += red[c + st];
        __syncthreads();
    }
    float sig = 1.f / (1.f + __expf(-(red[0] + attn_b[0])));
    base[(size_t)c * P_ + p] = v * sig;
}

// ---------------------------------------------------------------------------
extern "C" void kernel_launch(void* const* d_in, const int* in_sizes, int n_in,
                              void* d_out, int out_size, void* d_ws, size_t ws_size,
                              hipStream_t stream) {
    const float* x      = (const float*)d_in[0];
    const float* cw[3]  = {(const float*)d_in[1], (const float*)d_in[3], (const float*)d_in[5]};
    const float* cbi[3] = {(const float*)d_in[2], (const float*)d_in[4], (const float*)d_in[6]};
    const float* lnw    = (const float*)d_in[7];
    const float* lnb    = (const float*)d_in[8];
    const float* wci    = (const float*)d_in[9];
    const float* wcf    = (const float*)d_in[10];
    const float* wco    = (const float*)d_in[11];
    const float* attn_w = (const float*)d_in[12];
    const float* attn_b = (const float*)d_in[13];
    float* dout = (float*)d_out;

    const int ctot[3] = {CIN_ + HID_, 2 * HID_, 2 * HID_};
    int cpad[3];
    for (int l = 0; l < 3; ++l) cpad[l] = ((ctot[l] + 31) / 32) * 32;   // 160,256,256

    // -------- workspace carve-up (all offsets 256B aligned) --------
    char* ws = (char*)d_ws;
    size_t off = 0;
    auto carve = [&](size_t bytes) {
        size_t o = off; off += (bytes + 255) & ~(size_t)255; return (void*)(ws + o);
    };
    _Float16* wt[3];
    for (int l = 0; l < 3; ++l)
        wt[l] = (_Float16*)carve((size_t)25 * NGATE_ * cpad[l] * sizeof(_Float16));
    _Float16* comb  = (_Float16*)carve((size_t)B_ * P_ * 256 * sizeof(_Float16));
    float*    gates = (float*)carve((size_t)B_ * P_ * NGATE_ * sizeof(float));
    float*    cbuf  = (float*)carve((size_t)B_ * HID_ * P_ * sizeof(float));
    float*    pre   = (float*)carve((size_t)B_ * HID_ * P_ * sizeof(float));
    _Float16* seqA  = (_Float16*)carve((size_t)B_ * T_ * HID_ * P_ * sizeof(_Float16));
    _Float16* seqB  = (_Float16*)carve((size_t)B_ * T_ * HID_ * P_ * sizeof(_Float16));
    float*    stats = (float*)carve(B_ * 2 * sizeof(float));
    (void)ws_size; (void)in_sizes; (void)n_in; (void)out_size;

    // -------- weight repack (once per call; deterministic) --------
    for (int l = 0; l < 3; ++l) {
        int n = 25 * NGATE_ * cpad[l];
        wconv_kernel<<<(n + 255) / 256, 256, 0, stream>>>(cw[l], wt[l], ctot[l], cpad[l]);
    }

    const int npw = B_ * HID_ * P_;                 // pointwise element count
    const dim3 cgrid((P_ + 63) / 64, NGATE_ / 128, B_);

    for (int l = 0; l < 3; ++l) {
        int cin_l = (l == 0) ? CIN_ : HID_;
        const _Float16* seq_prev = (l == 0) ? nullptr : ((l == 1) ? seqA : seqB);
        _Float16* seq_cur        = (l == 1) ? seqB : seqA;   // l0->A, l1->B, l2->A
        float* dl = (l == 2) ? dout : nullptr;

        zero_kernel<<<(npw + 255) / 256, 256, 0, stream>>>(cbuf, npw);

        for (int t = 0; t < T_; ++t) {
            int npk = B_ * P_ * cpad[l];
            pack_comb_kernel<<<(npk + 255) / 256, 256, 0, stream>>>(
                (l == 0) ? x : nullptr, seq_prev, seq_cur, comb, cin_l, cpad[l], t);

            conv_wmma_kernel<<<cgrid, 128, 0, stream>>>(comb, wt[l], gates, cpad[l]);

            lstm_pointwise_kernel<<<(npw + 255) / 256, 256, 0, stream>>>(
                gates, cbi[l], wci + l * HID_, wcf + l * HID_, wco + l * HID_, cbuf, pre);

            stats_kernel<<<B_, 256, 0, stream>>>(pre, stats);

            norm_kernel<<<(npw + 255) / 256, 256, 0, stream>>>(
                pre, stats, lnw + (size_t)l * HID_ * P_, lnb + (size_t)l * HID_ * P_,
                seq_cur, dl, t);
        }
    }

    attn_kernel<<<B_ * P_, 128, 0, stream>>>(dout, attn_w, attn_b);
}
